// SeqDiscriminator_46557445488761
// MI455X (gfx1250) — matrix operand order repaired and checked by reading, not compiled, not run on hardware
//
#include <hip/hip_runtime.h>
#include <math.h>
#include <stdint.h>

typedef __bf16 bf16;
typedef __attribute__((ext_vector_type(16))) __bf16 v16bf;
typedef __attribute__((ext_vector_type(8)))  __bf16 v8bf;
typedef __attribute__((ext_vector_type(8)))  float  v8f;

#define D_MODEL 1024
#define GATES   3072
#define BATCH   32
#define SEQ     256
#define ROWS    (SEQ*BATCH)   /* 8192 */
#define LPAD    40            /* LDS row stride (elems): 80B, conflict-free */

union Frag16 { v16bf v; v8bf h[2]; };

// A-fragment (16x32 bf16): lane holds row M=lane%16, K=(lane/16)*16..+15
// (contiguous K order, matching the B fragment).
__device__ __forceinline__ v16bf ldA(const bf16* p) {
  Frag16 f; f.h[0] = *(const v8bf*)(p); f.h[1] = *(const v8bf*)(p + 8); return f.v;
}
// B-fragment (32x16 bf16): lane holds col N=lane%16, K=(lane/16)*16..+15,
// i.e. one contiguous 32-byte run of the (row-major W) row = output column.
__device__ __forceinline__ v16bf ldB(const bf16* p) {
  Frag16 f; f.h[0] = *(const v8bf*)(p); f.h[1] = *(const v8bf*)(p + 8); return f.v;
}
__device__ __forceinline__ v8f wmma_bf16(v16bf a, v16bf b, v8f c) {
  return __builtin_amdgcn_wmma_f32_16x16x32_bf16(false, a, false, b, (short)0, c,
                                                 false, false);
}

// ---- CDNA5 async global->LDS (ASYNCcnt-tracked, no VGPR staging) ----------
__device__ __forceinline__ void async_ld_b128(unsigned lds_addr, const void* gaddr) {
  asm volatile("global_load_async_to_lds_b128 %0, %1, off"
               :: "v"(lds_addr), "v"(gaddr) : "memory");
}
__device__ __forceinline__ void wait_async0() {
  asm volatile("s_wait_asynccnt 0x0" ::: "memory");
}
__device__ __forceinline__ void wait_async3() {
  asm volatile("s_wait_asynccnt 0x3" ::: "memory");
}
__device__ __forceinline__ unsigned lds_off(const void* p) {
  // generic LDS address: low 32 bits are the LDS byte offset
  return (unsigned)(uintptr_t)p;
}

// ---------------------------------------------------------------- utilities
__global__ void k_init(unsigned* bars) {
  if (threadIdx.x < 16) bars[threadIdx.x] = 0u;
}

__global__ void k_cvt(const float* __restrict__ in, bf16* __restrict__ out, int n) {
  int i = blockIdx.x * blockDim.x + threadIdx.x;
  int stride = gridDim.x * blockDim.x;
  for (; i < n; i += stride) out[i] = (bf16)in[i];
}

// fused input projection: x[t*32+b, :] = relu([cond@Wc^T+bc, input@Wi^T+bi])
__global__ void k_proj(const float* __restrict__ inp, const float* __restrict__ cond,
                       const float* __restrict__ Wc, const float* __restrict__ bc,
                       const float* __restrict__ Wi, const float* __restrict__ bi,
                       bf16* __restrict__ x0) {
  int gid = blockIdx.x * blockDim.x + threadIdx.x;     // ROWS*D_MODEL threads
  int col = gid & (D_MODEL - 1);
  int r   = gid >> 10;
  int t = r >> 5, b = r & 31;
  float acc;
  if (col < 512) {
    const float* cr = cond + ((size_t)b * SEQ + t) * 80;
    const float* wr = Wc + (size_t)col * 80;
    acc = bc[col];
    #pragma unroll 8
    for (int i = 0; i < 80; ++i) acc = fmaf(cr[i], wr[i], acc);
  } else {
    int c2 = col - 512;
    const float* xr = inp + ((size_t)b * SEQ + t) * 72;
    const float* wr = Wi + (size_t)c2 * 72;
    acc = bi[c2];
    #pragma unroll 8
    for (int i = 0; i < 72; ++i) acc = fmaf(xr[i], wr[i], acc);
  }
  x0[(size_t)r * D_MODEL + col] = (bf16)fmaxf(acc, 0.f);
}

// ------------------------------------------------ xg = A @ W^T + bias (WMMA)
// A: [8192,1024] bf16 rm, W: [3072,1024] bf16 rm, O: [8192,3072] bf16.
// block=256 (8 waves). WG tile M=128,N=64; wave tile 32x32 (2x2 WMMA tiles).
// A/B tiles double-buffered in LDS via GLOBAL_LOAD_ASYNC_TO_LDS_B128; k+1
// stage overlaps k-tile WMMAs; fragments read with ds_load_b128.
__global__ void __launch_bounds__(256)
k_gemm_xg(const bf16* __restrict__ A0, const bf16* __restrict__ A1,
          const bf16* __restrict__ W0, const bf16* __restrict__ W1,
          const float* __restrict__ b0, const float* __restrict__ b1,
          bf16* __restrict__ O0, bf16* __restrict__ O1) {
  __shared__ bf16 sA[2][128 * LPAD];   // 2 x 10KB
  __shared__ bf16 sB[2][64 * LPAD];    // 2 x 5KB
  const bf16* A = blockIdx.z ? A1 : A0;
  const bf16* W = blockIdx.z ? W1 : W0;
  const float* bias = blockIdx.z ? b1 : b0;
  bf16* O = blockIdx.z ? O1 : O0;

  const int tid = threadIdx.x;
  const int lane = tid & 31, wid = tid >> 5;
  const int wm = wid & 3, wn = wid >> 2;
  const int rowg = blockIdx.x * 128;          // global A row base
  const int colg = blockIdx.y * 64;           // global W row base (=out col)
  const int khA = (lane >> 4) << 4;
  const int khB = (lane >> 4) << 4;
  const int nlo = lane & 15;

  // per-thread staging chunks (16B each): A has 512 chunks, B has 256.
  const int ra0 = tid >> 2,          kca = (tid & 3) << 3;          // A chunk tid
  const int ra1 = (tid + 256) >> 2,  kca1 = kca;                    // A chunk tid+256
  const int rb  = tid >> 2,          kcb = (tid & 3) << 3;          // B chunk tid

  v8f c[2][2] = {};

  // stage k-tile 0
  async_ld_b128(lds_off(&sA[0][ra0 * LPAD + kca]),
                A + (size_t)(rowg + ra0) * D_MODEL + kca);
  async_ld_b128(lds_off(&sA[0][ra1 * LPAD + kca1]),
                A + (size_t)(rowg + ra1) * D_MODEL + kca1);
  async_ld_b128(lds_off(&sB[0][rb * LPAD + kcb]),
                W + (size_t)(colg + rb) * D_MODEL + kcb);

  for (int kt = 0; kt < D_MODEL / 32; ++kt) {
    const int cur = kt & 1;
    if (kt + 1 < D_MODEL / 32) {
      const int k1 = (kt + 1) * 32;
      async_ld_b128(lds_off(&sA[cur ^ 1][ra0 * LPAD + kca]),
                    A + (size_t)(rowg + ra0) * D_MODEL + k1 + kca);
      async_ld_b128(lds_off(&sA[cur ^ 1][ra1 * LPAD + kca1]),
                    A + (size_t)(rowg + ra1) * D_MODEL + k1 + kca1);
      async_ld_b128(lds_off(&sB[cur ^ 1][rb * LPAD + kcb]),
                    W + (size_t)(colg + rb) * D_MODEL + k1 + kcb);
      wait_async3();                 // batch kt done, batch kt+1 in flight
    } else {
      wait_async0();
    }
    __syncthreads();                 // tile kt visible to all waves

    v16bf a[2], bb[2];
    #pragma unroll
    for (int mi = 0; mi < 2; ++mi)
      a[mi] = ldA(&sA[cur][(wm * 32 + mi * 16 + nlo) * LPAD + khA]);
    #pragma unroll
    for (int ni = 0; ni < 2; ++ni)
      bb[ni] = ldB(&sB[cur][(wn * 32 + ni * 16 + nlo) * LPAD + khB]);
    #pragma unroll
    for (int mi = 0; mi < 2; ++mi)
      #pragma unroll
      for (int ni = 0; ni < 2; ++ni)
        c[mi][ni] = wmma_bf16(a[mi], bb[ni], c[mi][ni]);

    __syncthreads();                 // done reading buf before it is re-staged
  }

  const int vrow = (lane >> 4) << 3;
  const int row0 = rowg + wm * 32;
  const int col0 = colg + wn * 32;
  #pragma unroll
  for (int ni = 0; ni < 2; ++ni) {
    int col = col0 + ni * 16 + nlo;
    float bv = bias[col];
    #pragma unroll
    for (int mi = 0; mi < 2; ++mi)
      #pragma unroll
      for (int v = 0; v < 8; ++v) {
        int row = row0 + mi * 16 + v + vrow;
        O[(size_t)row * GATES + col] = (bf16)(c[mi][ni][v] + bv);
      }
  }
}

// ------------------------------------------------------- device grid barrier
__device__ __forceinline__ void gbar(unsigned* cnt, unsigned* gen, unsigned nwg) {
  __threadfence();
  __syncthreads();
  if (threadIdx.x == 0) {
    unsigned g = atomicAdd(gen, 0u);
    if (atomicAdd(cnt, 1u) == nwg - 1u) {
      atomicExch(cnt, 0u);
      __threadfence();
      atomicAdd(gen, 1u);
    } else {
      while (atomicAdd(gen, 0u) == g) __builtin_amdgcn_s_sleep(1);
    }
  }
  __syncthreads();
  __threadfence();
}

// --------------------------------------------- persistent GRU recurrence
// grid = (8, 2): 8 WGs per direction, blockIdx.y = direction.
// Each wave owns 16 hidden cols x 3 gates x both M-tiles (6 accumulators).
// h kept double-buffered in bf16 (WMMA A operand) + fp32 (gate update).
__global__ void __launch_bounds__(256)
k_gru(const bf16* __restrict__ xg0, const bf16* __restrict__ xg1,
      const bf16* __restrict__ Wh0, const bf16* __restrict__ Wh1,
      const float* __restrict__ bh0, const float* __restrict__ bh1,
      bf16* __restrict__ y0, bf16* __restrict__ y1,
      float* __restrict__ hf0, float* __restrict__ hf1,
      bf16* __restrict__ hb0, bf16* __restrict__ hb1,
      unsigned* __restrict__ bars) {
  const int dir = blockIdx.y;
  const bf16* xg = dir ? xg1 : xg0;
  const bf16* Wh = dir ? Wh1 : Wh0;
  const float* bh = dir ? bh1 : bh0;
  bf16* y  = dir ? y1 : y0;
  float* hf = dir ? hf1 : hf0;
  bf16* hb = dir ? hb1 : hb0;          // [2][32][1024] double buffer
  unsigned* cnt = bars + dir * 2;
  unsigned* gen = bars + dir * 2 + 1;
  const unsigned NWG = gridDim.x;

  const int lane = threadIdx.x & 31, wid = threadIdx.x >> 5;
  const int nlo = lane & 15;
  const int colL = blockIdx.x * 128 + wid * 16 + nlo;  // owned hidden column
  const int khA = (lane >> 4) << 4;
  const int khB = (lane >> 4) << 4;
  const int vrow = (lane >> 4) << 3;

  const float bhR = bh[colL];
  const float bhZ = bh[D_MODEL + colL];
  const float bhN = bh[2 * D_MODEL + colL];

  // zero this WG's slice of the hidden state (h0 = 0)
  {
    const int cbase = blockIdx.x * 128;
    for (int i = threadIdx.x; i < 128 * BATCH; i += blockDim.x) {
      int row = i >> 7, col = cbase + (i & 127);
      hf[row * D_MODEL + col] = 0.f;
      hb[row * D_MODEL + col] = (bf16)0.f;
      hb[BATCH * D_MODEL + row * D_MODEL + col] = (bf16)0.f;
    }
  }
  gbar(cnt, gen, NWG);

  for (int t = 0; t < SEQ; ++t) {
    const bf16* hrd = hb + (size_t)(t & 1) * BATCH * D_MODEL;
    bf16* hwr = hb + (size_t)((t + 1) & 1) * BATCH * D_MODEL;

    // prefetch this step's gate-precompute rows (independent of WMMA loop)
    #pragma unroll
    for (int m = 0; m < 2; ++m)
      __builtin_prefetch(xg + (size_t)(t * BATCH + m * 16 + vrow) * GATES + colL,
                         0, 1);

    v8f acc[2][3] = {};
    for (int k0 = 0; k0 < D_MODEL; k0 += 32) {
      v16bf a0 = ldA(hrd + (size_t)nlo * D_MODEL + k0 + khA);
      v16bf a1 = ldA(hrd + (size_t)(16 + nlo) * D_MODEL + k0 + khA);
      #pragma unroll
      for (int g = 0; g < 3; ++g) {
        v16bf bb = ldB(Wh + (size_t)(g * D_MODEL + colL) * D_MODEL + k0 + khB);
        acc[0][g] = wmma_bf16(a0, bb, acc[0][g]);
        acc[1][g] = wmma_bf16(a1, bb, acc[1][g]);
      }
    }

    #pragma unroll
    for (int m = 0; m < 2; ++m) {
      #pragma unroll
      for (int v = 0; v < 8; ++v) {
        int row = m * 16 + v + vrow;                   // batch index
        size_t xb = (size_t)(t * BATCH + row) * GATES + colL;
        float xr = (float)xg[xb];
        float xz = (float)xg[xb + D_MODEL];
        float xn = (float)xg[xb + 2 * D_MODEL];
        float hr = acc[m][0][v] + bhR;
        float hz = acc[m][1][v] + bhZ;
        float hn = acc[m][2][v] + bhN;
        float r = 1.f / (1.f + __expf(-(xr + hr)));
        float z = 1.f / (1.f + __expf(-(xz + hz)));
        float n = tanhf(xn + r * hn);
        float hp = hf[row * D_MODEL + colL];
        float hnw = (1.f - z) * n + z * hp;
        hf[row * D_MODEL + colL] = hnw;
        bf16 hv = (bf16)hnw;
        hwr[row * D_MODEL + colL] = hv;
        y[(size_t)(t * BATCH + row) * D_MODEL + colL] = hv;
      }
    }
    gbar(cnt, gen, NWG);
  }
}

// ------------------------------------------- output head: sigmoid((f+b)@Wo+bo)
__global__ void k_out(const bf16* __restrict__ yf, const bf16* __restrict__ yb,
                      const float* __restrict__ Wo, const float* __restrict__ bo,
                      float* __restrict__ scores) {
  int lane = threadIdx.x & 31, wid = threadIdx.x >> 5;
  int row = blockIdx.x * 8 + wid;
  float s = 0.f;
  for (int j = lane; j < D_MODEL; j += 32)
    s += ((float)yf[(size_t)row * D_MODEL + j] +
          (float)yb[(size_t)row * D_MODEL + j]) * Wo[j];
  #pragma unroll
  for (int off = 16; off; off >>= 1) s += __shfl_xor(s, off, 32);
  if (lane == 0) scores[row] = 1.f / (1.f + __expf(-(s + bo[0])));
}

__global__ void k_mean(const float* __restrict__ scores, float* __restrict__ out) {
  __shared__ float sm[256];
  float s = 0.f;
  for (int i = threadIdx.x; i < ROWS; i += 256) s += scores[i];
  sm[threadIdx.x] = s;
  __syncthreads();
  for (int off = 128; off; off >>= 1) {
    if (threadIdx.x < off) sm[threadIdx.x] += sm[threadIdx.x + off];
    __syncthreads();
  }
  if (threadIdx.x == 0) out[0] = sm[0] / (float)ROWS;
}

// ---------------------------------------------------------------------------
extern "C" void kernel_launch(void* const* d_in, const int* in_sizes, int n_in,
                              void* d_out, int out_size, void* d_ws, size_t ws_size,
                              hipStream_t stream) {
  (void)in_sizes; (void)n_in; (void)out_size; (void)ws_size;
  const float* inp   = (const float*)d_in[0];
  const float* cond  = (const float*)d_in[1];
  const float* Wc    = (const float*)d_in[2];
  const float* bc    = (const float*)d_in[3];
  const float* Wi    = (const float*)d_in[4];
  const float* bi    = (const float*)d_in[5];
  const float* Wih_f = (const float*)d_in[6];
  const float* Whh_f = (const float*)d_in[7];
  const float* bih_f = (const float*)d_in[8];
  const float* bhh_f = (const float*)d_in[9];
  const float* Wih_b = (const float*)d_in[10];
  const float* Whh_b = (const float*)d_in[11];
  const float* bih_b = (const float*)d_in[12];
  const float* bhh_b = (const float*)d_in[13];
  const float* Wo    = (const float*)d_in[14];
  const float* bo    = (const float*)d_in[15];
  float* out = (float*)d_out;

  char* ws = (char*)d_ws;
  size_t off = 0;
  auto alloc = [&](size_t bytes) {
    void* p = ws + off;
    off = (off + bytes + 255) & ~(size_t)255;
    return p;
  };
  const size_t WELEMS = (size_t)2 * GATES * D_MODEL;   // per weight array
  bf16* wih_f = (bf16*)alloc(WELEMS * 2);
  bf16* whh_f = (bf16*)alloc(WELEMS * 2);
  bf16* wih_b = (bf16*)alloc(WELEMS * 2);
  bf16* whh_b = (bf16*)alloc(WELEMS * 2);
  bf16* x0  = (bf16*)alloc((size_t)ROWS * D_MODEL * 2);
  bf16* yf  = (bf16*)alloc((size_t)ROWS * D_MODEL * 2);
  bf16* yb  = (bf16*)alloc((size_t)ROWS * D_MODEL * 2);
  bf16* xgf = (bf16*)alloc((size_t)ROWS * GATES * 2);
  bf16* xgb = (bf16*)alloc((size_t)ROWS * GATES * 2);
  float* hff = (float*)alloc((size_t)BATCH * D_MODEL * 4);
  float* hfb = (float*)alloc((size_t)BATCH * D_MODEL * 4);
  bf16* hbf = (bf16*)alloc((size_t)2 * BATCH * D_MODEL * 2);
  bf16* hbb = (bf16*)alloc((size_t)2 * BATCH * D_MODEL * 2);
  float* scores = (float*)alloc((size_t)ROWS * 4);
  unsigned* bars = (unsigned*)alloc(64);

  k_init<<<1, 64, 0, stream>>>(bars);
  const int ncvt = (int)WELEMS;
  k_cvt<<<2048, 256, 0, stream>>>(Wih_f, wih_f, ncvt);
  k_cvt<<<2048, 256, 0, stream>>>(Whh_f, whh_f, ncvt);
  k_cvt<<<2048, 256, 0, stream>>>(Wih_b, wih_b, ncvt);
  k_cvt<<<2048, 256, 0, stream>>>(Whh_b, whh_b, ncvt);
  k_proj<<<(ROWS * D_MODEL) / 256, 256, 0, stream>>>(inp, cond, Wc, bc, Wi, bi, x0);

  dim3 gg(ROWS / 128, GATES / 64, 2);
  // layer 0
  k_gemm_xg<<<gg, 256, 0, stream>>>(x0, x0, wih_f, wih_b, bih_f, bih_b, xgf, xgb);
  k_gru<<<dim3(8, 2), 256, 0, stream>>>(xgf, xgb, whh_f, whh_b, bhh_f, bhh_b,
                                        yf, yb, hff, hfb, hbf, hbb, bars);
  // layer 1
  const size_t LW = (size_t)GATES * D_MODEL;
  k_gemm_xg<<<gg, 256, 0, stream>>>(yf, yb, wih_f + LW, wih_b + LW,
                                    bih_f + GATES, bih_b + GATES, xgf, xgb);
  k_gru<<<dim3(8, 2), 256, 0, stream>>>(xgf, xgb, whh_f + LW, whh_b + LW,
                                        bhh_f + GATES, bhh_b + GATES,
                                        yf, yb, hff, hfb, hbf, hbb, bars);

  k_out<<<ROWS / 8, 256, 0, stream>>>(yf, yb, Wo, bo, scores);
  k_mean<<<1, 256, 0, stream>>>(scores, out);
}
